// TransformerXLModel_68771016344130
// MI455X (gfx1250) — matrix operand order repaired and checked by reading, not compiled, not run on hardware
//
#include <hip/hip_runtime.h>

typedef __attribute__((ext_vector_type(16))) __bf16 v16bf;
typedef __attribute__((ext_vector_type(8)))  float  v8f;
typedef __attribute__((ext_vector_type(4)))  unsigned int v4u;
typedef __attribute__((ext_vector_type(8)))  int    v8i;
typedef __attribute__((ext_vector_type(4)))  int    v4i;

// ---- fixed problem dims from the reference ----
constexpr int LL = 6;
constexpr int BB = 4;
constexpr int QQ = 512;
constexpr int MMEM = 512;
constexpr int DD = 1024;
constexpr int HH = 16;
constexpr int FF = 4096;
constexpr int SS = 64;
constexpr int RR = QQ + MMEM;            // 1024
constexpr float NEGINF = -1000000000.0f;

// ================= generic batched WMMA GEMM ====================
// C[m,n] = sum_k A[m,k]*B[k,n] (+bias[n]) (opt ReLU), bf16 math, f32 acc.
// A is always bf16 with unit k-stride (16B-aligned rows).
// TDMB=true: B is fp32 row-major (sBn==1); tiles streamed by the Tensor
//            Data Mover into padded LDS (65-dword rows, conflict-free).
// Per-z offsets: (z/zdiv)*outer + (z%zdiv)*inner for A,B,C.
constexpr int BMT = 128, BNT = 64, BKT = 32;

template <typename TB, typename TC, bool RELU, bool TDMB>
__global__ __launch_bounds__(256)
void gemm_wmma_k(const __bf16* __restrict__ Ag, const TB* __restrict__ Bg,
                 TC* __restrict__ Cg, const float* __restrict__ bias,
                 int Md, int Nd, int Kd,
                 long sAm, long sBk, long sBn, long sCm,
                 int zdiv, long aO, long aI, long bO, long bI, long cO, long cI)
{
  const int z = blockIdx.z;
  Ag += (long)(z / zdiv) * aO + (long)(z % zdiv) * aI;
  Bg += (long)(z / zdiv) * bO + (long)(z % zdiv) * bI;
  Cg += (long)(z / zdiv) * cO + (long)(z % zdiv) * cI;
  const int m0 = blockIdx.y * BMT;
  const int n0 = blockIdx.x * BNT;

  __shared__ __bf16 As[BMT][BKT + 8];      // +8 keeps 16B store alignment
  __shared__ __bf16 Bh[BKT][BNT + 2];      // generic (strided/bf16) B path
  __shared__ float  Bf[BKT * 65];          // TDM path: 64 + 1 pad per row

  const int tid  = threadIdx.x;
  const int lane = tid & 31;
  const int wave = tid >> 5;               // 8 waves: wave w owns rows [16w,16w+16)

  v8f acc[4];
#pragma unroll
  for (int i = 0; i < 4; i++)
#pragma unroll
    for (int j = 0; j < 8; j++) acc[i][j] = 0.0f;

  unsigned long long gb = 0;
  unsigned ldsb = 0;
  if constexpr (TDMB) {
    gb   = (unsigned long long)(uintptr_t)(Bg + (long)n0);
    ldsb = (unsigned)(uintptr_t)&Bf[0];
  }

  for (int k0 = 0; k0 < Kd; k0 += BKT) {
    // ---- stage A tile (BMT x BKT bf16) with 128-bit loads ----
#pragma unroll
    for (int e = tid; e < BMT * BKT / 8; e += 256) {
      int r = e >> 2, c8 = (e & 3) << 3;
      *(uint4*)&As[r][c8] =
          *(const uint4*)(Ag + (long)(m0 + r) * sAm + (k0 + c8));
    }
    // ---- stage B tile ----
    if constexpr (TDMB) {
      if (wave == 0) {
        // Tensor DMA descriptor (D#): 2-D fp32 tile, 64 x 32 rows,
        // LDS padding: +1 dword per 64 dwords -> 65-dword LDS rows.
        unsigned long long ga =
            gb + (unsigned long long)k0 * (unsigned long long)sBk * 4ull;
        v4u g0;
        g0[0] = 1u;                                   // count=1, user mode
        g0[1] = ldsb;                                 // lds_addr
        g0[2] = (unsigned)ga;                         // global_addr[31:0]
        g0[3] = (unsigned)((ga >> 32) & 0x1ffffffu)   // global_addr[56:32]
                | 0x80000000u;                        // type=2 ("image")
        unsigned td0 = (unsigned)sBk;                 // tensor dim0 (>= Nd)
        unsigned td1 = (unsigned)Kd;                  // tensor dim1
        v8i g1;
        g1[0] = (int)(0x20000u       // data_size=2 (4 bytes)
                      | 0x100000u    // pad_enable
                      | (5u << 22)); // pad_interval: every 64 dwords (+1 dw)
        g1[1] = (int)((td0 & 0xffffu) << 16);               // dim0[15:0]
        g1[2] = (int)((td0 >> 16) | ((td1 & 0xffffu) << 16));
        g1[3] = (int)((td1 >> 16) | (64u << 16));           // tile_dim0=64
        g1[4] = 32;                                         // tile_dim1=32
        g1[5] = (int)(unsigned)sBk;                         // dim0 stride
        g1[6] = 0;
        g1[7] = 0;
        v4i z4;
        z4[0] = z4[1] = z4[2] = z4[3] = 0;
        v8i z8;
#pragma unroll
        for (int q8 = 0; q8 < 8; q8++) z8[q8] = 0;
        __builtin_amdgcn_tensor_load_to_lds(g0, g1, z4, z4, z8, 0);
        __builtin_amdgcn_s_wait_tensorcnt(0);
      }
    } else {
      for (int e = tid; e < BKT * BNT; e += 256) {
        int r = e / BNT, c = e % BNT;
        Bh[r][c] = (__bf16)(float)Bg[(long)(k0 + r) * sBk +
                                     (long)(n0 + c) * sBn];
      }
    }
    __syncthreads();

    // A fragment 16x32 bf16 (ISA layout): lanes 0-15 row M=l, halfs
    // 0..7 -> K=0..7, 8..15 -> K=16..23; lanes 16-31 same row, K +8.
    v16bf af;
    const int rowA = wave * 16 + (lane & 15);
    const int kgrp = lane >> 4;
#pragma unroll
    for (int h = 0; h < 16; h++) {
      int kk = (h & 7) + 8 * (((h >> 3) << 1) + kgrp);
      af[h] = As[rowA][kk];
    }
#pragma unroll
    for (int nt = 0; nt < 4; nt++) {
      // B fragment 32x16: lane%16 = N column, halfs = K rows (same K order as A)
      v16bf bfr;
#pragma unroll
      for (int h = 0; h < 16; h++) {
        int kb = (h & 7) + 8 * (((h >> 3) << 1) + kgrp);
        if constexpr (TDMB)
          bfr[h] = (__bf16)Bf[kb * 65 + nt * 16 + (lane & 15)];
        else
          bfr[h] = Bh[kb][nt * 16 + (lane & 15)];
      }
      acc[nt] = __builtin_amdgcn_wmma_f32_16x16x32_bf16(
          false, af, false, bfr, (short)0, acc[nt], false, false);
    }
    __syncthreads();
  }

  // C layout: VGPR i -> row = i + 8*(lane/16), col = lane%16
  const int ccol  = lane & 15;
  const int rbase = m0 + wave * 16 + 8 * (lane >> 4);
#pragma unroll
  for (int nt = 0; nt < 4; nt++) {
    const int ncol = n0 + nt * 16 + ccol;
    const float bv = bias ? bias[ncol] : 0.0f;
#pragma unroll
    for (int i = 0; i < 8; i++) {
      float v = acc[nt][i] + bv;
      if (RELU) v = v > 0.0f ? v : 0.0f;
      Cg[(long)(rbase + i) * sCm + ncol] = (TC)v;
    }
  }
}

template <typename TB, typename TC, bool RELU, bool TDMB>
static void launch_gemm(hipStream_t s, const void* A, const void* Bm, void* C,
                        const float* bias, int Md, int Nd, int Kd,
                        long sAm, long sBk, long sBn, long sCm,
                        int zb, int zdiv, long aO, long aI, long bO, long bI,
                        long cO, long cI)
{
  dim3 g(Nd / BNT, Md / BMT, zb), blk(256);
  hipLaunchKernelGGL((gemm_wmma_k<TB, TC, RELU, TDMB>), g, blk, 0, s,
                     (const __bf16*)A, (const TB*)Bm, (TC*)C, bias, Md, Nd, Kd,
                     sAm, sBk, sBn, sCm, zdiv, aO, aI, bO, bI, cO, cI);
}

// ======================= support kernels ========================
__global__ __launch_bounds__(256)
void embed_k(const int* __restrict__ tok, const float* __restrict__ tab,
             float* __restrict__ xf, __bf16* __restrict__ xb)
{
  const long row = blockIdx.x;                  // b*QQ + q
  const float* e = tab + (long)tok[row] * DD;
  for (int d = threadIdx.x; d < DD; d += 256) {
    float v = e[d] * 32.0f;                     // sqrt(1024)
    xf[row * DD + d] = v;
    xb[row * DD + d] = (__bf16)v;
  }
}

__global__ __launch_bounds__(256)
void posenc_k(__bf16* __restrict__ pb)
{
  const int r = blockIdx.x;
  const float pos = (float)(RR - 1 - r);
  for (int d = threadIdx.x; d < DD; d += 256) {
    int i = d & (DD / 2 - 1);
    float invf = powf(10000.0f, -2.0f * (float)i / (float)DD);
    float ang = pos * invf;
    pb[(long)r * DD + d] = (__bf16)((d < DD / 2) ? sinf(ang) : cosf(ang));
  }
}

__global__ __launch_bounds__(256)
void refpack_k(const float* __restrict__ mem, const float* __restrict__ xf,
               __bf16* __restrict__ ref)
{
  const long row = blockIdx.x;                  // b*RR + r
  const long b = row / RR;
  const int r = (int)(row % RR);
  const float* src = (r < MMEM) ? (mem + (b * MMEM + r) * (long)DD)
                                : (xf + (b * QQ + (r - MMEM)) * (long)DD);
  __bf16* dst = ref + row * (long)DD;
  for (int d = threadIdx.x; d < DD; d += 256) dst[d] = (__bf16)src[d];
}

__global__ __launch_bounds__(256)
void qbias_k(const float* __restrict__ q, const float* __restrict__ cb,
             const float* __restrict__ pb, __bf16* __restrict__ qc,
             __bf16* __restrict__ qp)
{
  const long idx = (long)blockIdx.x * 256 + threadIdx.x;
  const int col = (int)(idx & (DD - 1));        // (h,s) flattened
  float v = q[idx];
  qc[idx] = (__bf16)(v + cb[col]);
  qp[idx] = (__bf16)(v + pb[col]);
}

// logits = (content + rel_shift(positions))/sqrt(S) + mask*NEG_INF; softmax.
__global__ __launch_bounds__(256)
void attn_softmax_k(const float* __restrict__ SC, const float* __restrict__ SP,
                    __bf16* __restrict__ W)
{
  const int tid = threadIdx.x;
  const long row = blockIdx.x;                  // (b*HH + h)*QQ + i
  const int i = (int)(row % QQ);
  const long bh = row / QQ;
  const float* crow = SC + row * RR;
  const float* pbase = SP + bh * (long)QQ * RR;
  __shared__ float red[256];

  float lg[4];
  float lmax = -3.4e38f;
#pragma unroll
  for (int kk = 0; kk < 4; kk++) {
    int j = kk * 256 + tid;
    long t = QQ + (long)i * RR + j;             // rel_shift index math
    int i2 = (int)(t / (RR + 1));
    int j2 = (int)(t - (long)i2 * (RR + 1));
    float p = (j2 == 0) ? 0.0f : pbase[(long)i2 * RR + (j2 - 1)];
    float v = (crow[j] + p) * 0.125f;           // 1/sqrt(64)
    if (j > i + MMEM) v += NEGINF;
    lg[kk] = v;
    lmax = fmaxf(lmax, v);
  }
  red[tid] = lmax; __syncthreads();
  for (int s2 = 128; s2 > 0; s2 >>= 1) {
    if (tid < s2) red[tid] = fmaxf(red[tid], red[tid + s2]);
    __syncthreads();
  }
  float mx = red[0]; __syncthreads();
  float lsum = 0.0f;
#pragma unroll
  for (int kk = 0; kk < 4; kk++) { lg[kk] = expf(lg[kk] - mx); lsum += lg[kk]; }
  red[tid] = lsum; __syncthreads();
  for (int s2 = 128; s2 > 0; s2 >>= 1) {
    if (tid < s2) red[tid] += red[tid + s2];
    __syncthreads();
  }
  float inv = 1.0f / red[0];
#pragma unroll
  for (int kk = 0; kk < 4; kk++) {
    int j = kk * 256 + tid;
    W[row * RR + j] = (__bf16)(lg[kk] * inv);
  }
}

// out = LayerNorm(a + r) * g + b ; writes fp32 + bf16 copies
__global__ __launch_bounds__(256)
void resid_ln_k(const float* __restrict__ a, const float* __restrict__ r,
                const float* __restrict__ g, const float* __restrict__ bta,
                float* __restrict__ of, __bf16* __restrict__ ob)
{
  const int tid = threadIdx.x;
  const long row = blockIdx.x;
  const float* ar = a + row * DD;
  const float* rr2 = r + row * DD;
  __shared__ float rs[256], rq[256];
  float s = 0.0f, q2 = 0.0f;
  for (int d = tid; d < DD; d += 256) {
    float v = ar[d] + rr2[d];
    s += v; q2 += v * v;
  }
  rs[tid] = s; rq[tid] = q2; __syncthreads();
  for (int st = 128; st > 0; st >>= 1) {
    if (tid < st) { rs[tid] += rs[tid + st]; rq[tid] += rq[tid + st]; }
    __syncthreads();
  }
  float mu = rs[0] * (1.0f / DD);
  float var = rq[0] * (1.0f / DD) - mu * mu;
  float inv = rsqrtf(var + 1e-12f);
  for (int d = tid; d < DD; d += 256) {
    float v = (ar[d] + rr2[d] - mu) * inv * g[d] + bta[d];
    of[row * DD + d] = v;
    ob[row * DD + d] = (__bf16)v;
  }
}

// ========================= host driver ==========================
extern "C" void kernel_launch(void* const* d_in, const int* in_sizes, int n_in,
                              void* d_out, int out_size, void* d_ws, size_t ws_size,
                              hipStream_t stream)
{
  (void)in_sizes; (void)n_in; (void)out_size; (void)ws_size;
  const int*   inputs   = (const int*)d_in[0];
  const float* memories = (const float*)d_in[1];
  const float* embed    = (const float*)d_in[2];
  const float* wq       = (const float*)d_in[3];
  const float* wkc      = (const float*)d_in[4];
  const float* wkp      = (const float*)d_in[5];
  const float* wv       = (const float*)d_in[6];
  const float* wo       = (const float*)d_in[7];
  const float* cbias    = (const float*)d_in[8];
  const float* pbias    = (const float*)d_in[9];
  const float* w1       = (const float*)d_in[10];
  const float* b1       = (const float*)d_in[11];
  const float* w2       = (const float*)d_in[12];
  const float* b2       = (const float*)d_in[13];
  const float* ln1g     = (const float*)d_in[14];
  const float* ln1b     = (const float*)d_in[15];
  const float* ln2g     = (const float*)d_in[16];
  const float* ln2b     = (const float*)d_in[17];

  char* ws = (char*)d_ws;
  size_t off = 0;
  auto alloc = [&](size_t bytes) -> void* {
    void* p = ws + off;
    off = (off + bytes + 255) & ~(size_t)255;
    return p;
  };
  const long NBQ = (long)BB * QQ;               // 2048 token rows
  const long NBR = (long)BB * RR;               // 4096 ref rows

  float*  xf   = (float*)alloc(NBQ * DD * 4);
  __bf16* xb   = (__bf16*)alloc(NBQ * DD * 2);
  __bf16* posb = (__bf16*)alloc((long)RR * DD * 2);
  __bf16* refb = (__bf16*)alloc(NBR * DD * 2);
  float*  qf   = (float*)alloc(NBQ * DD * 4);
  __bf16* qcb  = (__bf16*)alloc(NBQ * DD * 2);
  __bf16* qpb  = (__bf16*)alloc(NBQ * DD * 2);
  __bf16* kcb  = (__bf16*)alloc(NBR * DD * 2);
  __bf16* kpb  = (__bf16*)alloc((long)RR * DD * 2);
  __bf16* vb   = (__bf16*)alloc(NBR * DD * 2);
  float*  S1   = (float*)alloc((long)BB * HH * QQ * RR * 4);
  float*  S2   = (float*)alloc((long)BB * HH * QQ * RR * 4);
  __bf16* Wb   = (__bf16*)alloc((long)BB * HH * QQ * RR * 2);
  __bf16* attb = (__bf16*)alloc(NBQ * DD * 2);
  float*  aof  = (float*)alloc(NBQ * DD * 4);
  float*  yf   = (float*)alloc(NBQ * DD * 4);
  __bf16* yb   = (__bf16*)alloc(NBQ * DD * 2);
  __bf16* hb   = (__bf16*)alloc(NBQ * (long)FF * 2);
  float*  ofb  = (float*)alloc(NBQ * DD * 4);

  hipLaunchKernelGGL(embed_k, dim3(NBQ), dim3(256), 0, stream, inputs, embed, xf, xb);
  hipLaunchKernelGGL(posenc_k, dim3(RR), dim3(256), 0, stream, posb);

  for (int l = 0; l < LL; l++) {
    const float* wq_l  = wq  + (long)l * DD * DD;
    const float* wkc_l = wkc + (long)l * DD * DD;
    const float* wkp_l = wkp + (long)l * DD * DD;
    const float* wv_l  = wv  + (long)l * DD * DD;
    const float* wo_l  = wo  + (long)l * DD * DD;
    const float* mem_l = memories + (long)l * BB * MMEM * DD;

    // ref = concat(mem, x) -> bf16
    hipLaunchKernelGGL(refpack_k, dim3(NBR), dim3(256), 0, stream, mem_l, xf, refb);

    // projections: A activations bf16, B weights fp32 streamed via TDM
    launch_gemm<float, float, false, true>(stream, xb, wq_l, qf, nullptr,
        NBQ, DD, DD, DD, DD, 1, DD, 1, 1, 0, 0, 0, 0, 0, 0);
    launch_gemm<float, __bf16, false, true>(stream, refb, wkc_l, kcb, nullptr,
        NBR, DD, DD, DD, DD, 1, DD, 1, 1, 0, 0, 0, 0, 0, 0);
    launch_gemm<float, __bf16, false, true>(stream, posb, wkp_l, kpb, nullptr,
        RR, DD, DD, DD, DD, 1, DD, 1, 1, 0, 0, 0, 0, 0, 0);
    launch_gemm<float, __bf16, false, true>(stream, refb, wv_l, vb, nullptr,
        NBR, DD, DD, DD, DD, 1, DD, 1, 1, 0, 0, 0, 0, 0, 0);

    // q + content_bias / position_bias (bias is per flattened (h,s) column)
    hipLaunchKernelGGL(qbias_k, dim3((unsigned)(NBQ * DD / 256)), dim3(256), 0, stream,
                       qf, cbias + (long)l * DD, pbias + (long)l * DD, qcb, qpb);

    // content: (Q,S)x(S,R) per z=(b,h);  B[s][r]=kc[b,r,h,s]
    launch_gemm<__bf16, float, false, false>(stream, qcb, kcb, S1, nullptr,
        QQ, RR, SS,
        (long)HH * SS,                    // A row stride
        1, (long)HH * SS,                 // B strides (k,n)
        RR,                               // C row stride
        BB * HH, HH,
        (long)QQ * DD, SS,                // A per-b / per-h
        (long)RR * DD, SS,                // B per-b / per-h
        (long)HH * QQ * RR, (long)QQ * RR);
    // positions: B[s][r]=kp[r,h,s] (no batch dim on B's b)
    launch_gemm<__bf16, float, false, false>(stream, qpb, kpb, S2, nullptr,
        QQ, RR, SS,
        (long)HH * SS,
        1, (long)HH * SS,
        RR,
        BB * HH, HH,
        (long)QQ * DD, SS,
        0, SS,
        (long)HH * QQ * RR, (long)QQ * RR);

    // rel_shift + scale + causal mask + softmax -> bf16 weights
    hipLaunchKernelGGL(attn_softmax_k, dim3((unsigned)((long)BB * HH * QQ)),
                       dim3(256), 0, stream, S1, S2, Wb);

    // att = W x V, scattered straight into (B,Q,H,S) bf16
    launch_gemm<__bf16, __bf16, false, false>(stream, Wb, vb, attb, nullptr,
        QQ, SS, RR,
        RR,
        (long)HH * SS, 1,
        (long)HH * SS,
        BB * HH, HH,
        (long)HH * QQ * RR, (long)QQ * RR,
        (long)RR * DD, SS,
        (long)QQ * DD, SS);

    // output projection (TDM weight streaming)
    launch_gemm<float, float, false, true>(stream, attb, wo_l, aof, nullptr,
        NBQ, DD, DD, DD, DD, 1, DD, 1, 1, 0, 0, 0, 0, 0, 0);

    // y = LN(att_out + x)
    hipLaunchKernelGGL(resid_ln_k, dim3((unsigned)NBQ), dim3(256), 0, stream,
                       aof, xf, ln1g + (long)l * DD, ln1b + (long)l * DD, yf, yb);

    // FFN: relu(y@w1+b1) -> bf16, then @w2+b2 -> fp32 (TDM weight streaming)
    launch_gemm<float, __bf16, true, true>(stream, yb, w1 + (long)l * DD * FF, hb,
        b1 + (long)l * FF,
        NBQ, FF, DD, DD, FF, 1, FF, 1, 1, 0, 0, 0, 0, 0, 0);
    launch_gemm<float, float, false, true>(stream, hb, w2 + (long)l * FF * DD, ofb,
        b2 + (long)l * DD,
        NBQ, DD, FF, FF, DD, 1, DD, 1, 1, 0, 0, 0, 0, 0, 0);

    // x = LN(o + y)
    hipLaunchKernelGGL(resid_ln_k, dim3((unsigned)NBQ), dim3(256), 0, stream,
                       ofb, yf, ln2g + (long)l * DD, ln2b + (long)l * DD, xf, xb);
  }

  (void)hipMemcpyAsync(d_out, xf, NBQ * (long)DD * sizeof(float),
                       hipMemcpyDeviceToDevice, stream);
}
